// PhysicallyRecurrentStudentPolicy_13426067767932
// MI455X (gfx1250) — matrix operand order repaired and checked by reading, not compiled, not run on hardware
//
#include <hip/hip_runtime.h>
#include <cstdint>

#define EPS 1e-5f

typedef __attribute__((ext_vector_type(16))) __bf16 bf16x16;
typedef __attribute__((ext_vector_type(8)))  float  f32x8;

// ---------------- input index map (setup_inputs dict order, depth-first) ---
enum {
  IN_PC = 0, IN_PROP = 1,
  IN_VW1 = 2, IN_VB1, IN_BN1_G, IN_BN1_B, IN_BN1_M, IN_BN1_V,
  IN_VW2, IN_VB2, IN_BN2_G, IN_BN2_B, IN_BN2_M, IN_BN2_V,
  IN_VW3, IN_VB3, IN_BN3_G, IN_BN3_B, IN_BN3_M, IN_BN3_V,
  IN_PM_W1, IN_PM_B1, IN_PM_W2, IN_PM_B2, IN_PM_LN_G, IN_PM_LN_B,
  IN_H_INIT, IN_QP_W, IN_QP_B,
  IN_CA_WIN, IN_CA_BIN, IN_CA_WOUT, IN_CA_BOUT,
  IN_AN_G, IN_AN_B,
  IN_R_W1, IN_R_B1, IN_R_W2, IN_R_B2, IN_R_LN_G, IN_R_LN_B,
  IN_POS_EMB,
  IN_LH_W1, IN_LH_B1, IN_LH_W2, IN_LH_B2,
  IN_TD_W1, IN_TD_B1, IN_TD_W2, IN_TD_B2,
  IN_L0 = 50          // + L*12 : ln1_g, ln1_b, sa_win, sa_bin, sa_wout, sa_bout,
                      //          ln2_g, ln2_b, ff_w1, ff_b1, ff_w2, ff_b2
};

// ---------------- workspace layout (byte offsets) --------------------------
#define WS_W2BF   0u            // 8192  bf16  (BN-folded vw2)
#define WS_W3BF   16384u        // 32768 bf16  (BN-folded vw3)   [contig async region 0..81920)
#define WS_W1F    81920u        // 192 f32
#define WS_B1F    82688u        // 64  f32
#define WS_B2F    82944u        // 128 f32
#define WS_B3F    83456u        // 256 f32
#define WS_VIS    84480u        // 65536 f32  (B*T=256 rows x 256)
#define WS_PROPO  346624u       // 65536 f32
#define WS_X      608768u       // 65536 f32  transformer activations
#define WS_HFIN   870912u       // 8192  f32

// ---------------- helpers ---------------------------------------------------
__device__ __forceinline__ f32x8 wmma_bf16(bf16x16 a, bf16x16 b, f32x8 c) {
  return __builtin_amdgcn_wmma_f32_16x16x32_bf16(false, a, false, b, (short)0, c,
                                                 false, false);
}

// K offset inside a 32-wide K block for A-fragment element j (ISA 7.12.2)
__device__ __forceinline__ int a_k(int j, int hl) {
  int v = j >> 1, p = j & 1;
  return ((v < 4) ? (2 * v + 8 * hl) : (16 + 2 * (v - 4) + 8 * hl)) + p;
}

// position of K (within its 32-block) in fragment order: segment = hl that owns
// this K, offset = element index j.  perm(k) = 16*k[3] + (k&7) + 8*k[4]
__device__ __forceinline__ int a_perm(int k) {
  return (((k >> 3) & 1) << 4) + (k & 7) + (((k >> 4) & 1) << 3);
}

// 16 contiguous bf16 halves
__device__ __forceinline__ bf16x16 load_frag_contig(const __bf16* p) {
  bf16x16 f;
#pragma unroll
  for (int j = 0; j < 16; ++j) f[j] = p[j];
  return f;
}

// float atomic max via sign-split int trick (lowers to global_atomic_max/min)
__device__ __forceinline__ void atomicMaxF(float* a, float v) {
  if (v >= 0.f) atomicMax((int*)a, __float_as_int(v));
  else          atomicMin((unsigned int*)a, __float_as_uint(v));
}

__device__ __forceinline__ float block_sum256(float v, float* red) {
  red[threadIdx.x] = v;
  __syncthreads();
#pragma unroll
  for (int s = 128; s > 0; s >>= 1) {
    if ((int)threadIdx.x < s) red[threadIdx.x] += red[threadIdx.x + s];
    __syncthreads();
  }
  float r = red[0];
  __syncthreads();
  return r;
}

// ============================================================================
// Kernel 1: fold BN into weights (bf16), init vis to -inf-ish
// ============================================================================
__global__ __launch_bounds__(256) void prep_kernel(
    const float* vw1, const float* vb1, const float* g1, const float* bb1,
    const float* m1, const float* v1,
    const float* vw2, const float* vb2, const float* g2, const float* bb2,
    const float* m2, const float* v2,
    const float* vw3, const float* vb3, const float* g3, const float* bb3,
    const float* m3, const float* v3,
    __bf16* W2bf, __bf16* W3bf, float* W1f, float* b1f, float* b2f, float* b3f,
    float* vis) {
  int i = blockIdx.x * 256 + threadIdx.x;
  if (i < 65536) vis[i] = -3.0e38f;
  if (i < 32768) { int n = i >> 7; float s = g3[n] * rsqrtf(v3[n] + EPS); W3bf[i] = (__bf16)(vw3[i] * s); }
  if (i < 8192)  { int n = i >> 6; float s = g2[n] * rsqrtf(v2[n] + EPS); W2bf[i] = (__bf16)(vw2[i] * s); }
  if (i < 192)   { int n = i / 3;  float s = g1[n] * rsqrtf(v1[n] + EPS); W1f[i]  = vw1[i] * s; }
  if (i < 256)   { float s = g3[i] * rsqrtf(v3[i] + EPS); b3f[i] = s * (vb3[i] - m3[i]) + bb3[i]; }
  if (i < 128)   { float s = g2[i] * rsqrtf(v2[i] + EPS); b2f[i] = s * (vb2[i] - m2[i]) + bb2[i]; }
  if (i < 64)    { float s = g1[i] * rsqrtf(v1[i] + EPS); b1f[i] = s * (vb1[i] - m1[i]) + bb1[i]; }
}

// ============================================================================
// Kernel 2: fused PointNet encoder (WMMA bf16) + max-pool
//   grid (256 bt, 4 chunks) x 256 threads (8 waves); each wave: 16-pt tiles
// ============================================================================
__global__ __launch_bounds__(256) void pointnet_kernel(
    const float* __restrict__ pc, const __bf16* __restrict__ wbf,
    const float* __restrict__ W1f, const float* __restrict__ b1f,
    const float* __restrict__ b2f, const float* __restrict__ b3f,
    float* __restrict__ vis) {
  __shared__ __bf16 Ws[8192 + 32768];           // W2 (128x64) || W3 (256x128)
  __shared__ float  W1s[192], b1s[64], b2s[128], b3s[256];
  __shared__ __bf16 h2t[8][16 * 128];           // per-wave layer-2 tile (natural cols)

  const int tid  = threadIdx.x;
  const int lane = tid & 31, wave = tid >> 5;
  const int hl = lane >> 4, nl = lane & 15;

  // ---- stage folded weights into LDS with async loads (ASYNCcnt path) ----
  {
    const char* gsrc = (const char*)wbf;
    for (int off = tid * 16; off < 81920; off += 256 * 16) {
      unsigned ldsa = (unsigned)(unsigned long long)(const void*)((const char*)Ws + off);
      const char* g = gsrc + off;
      asm volatile("global_load_async_to_lds_b128 %0, %1, off"
                   :: "v"(ldsa), "v"(g) : "memory");
    }
  }
  for (int i = tid; i < 192; i += 256) W1s[i] = W1f[i];
  if (tid < 64)  b1s[tid] = b1f[tid];
  if (tid < 128) b2s[tid] = b2f[tid];
  b3s[tid] = b3f[tid];
  asm volatile("s_wait_asynccnt 0x0" ::: "memory");
  __syncthreads();

  const __bf16* W2s = Ws;
  const __bf16* W3s = Ws + 8192;
  __bf16* h2 = &h2t[wave][0];

  const int bt = blockIdx.x;
  const float* P0 = pc + (size_t)bt * (2048 * 3);
  const int m = nl;  // this lane's A-matrix row (point within tile)

  for (int it = 0; it < 4; ++it) {
    const int p0 = blockIdx.y * 512 + it * 128 + wave * 16;
    const float* P = P0 + (size_t)(p0 + m) * 3;
    const float px = P[0], py = P[1], pz = P[2];
    if (it < 3)  // speculative prefetch of next tile's points
      __builtin_prefetch(P0 + (size_t)(p0 + 128 + m) * 3, 0, 1);

    // ---- layer 1 (K=3) computed straight into layer-2 A fragments ----
    bf16x16 A2[2];
#pragma unroll
    for (int kt = 0; kt < 2; ++kt)
#pragma unroll
      for (int j = 0; j < 16; ++j) {
        int c = kt * 32 + 16 * hl + j;
        float v = fmaf(W1s[c * 3 + 0], px,
                  fmaf(W1s[c * 3 + 1], py,
                  fmaf(W1s[c * 3 + 2], pz, b1s[c])));
        A2[kt][j] = (__bf16)fmaxf(v, 0.f);
      }

    // ---- layer 2: 16x64 @ 64x128 -> 16 WMMAs (B frags grouped per K-step) --
    f32x8 acc[8];
#pragma unroll
    for (int nt = 0; nt < 8; ++nt) acc[nt] = (f32x8)0.f;
#pragma unroll
    for (int kt = 0; kt < 2; ++kt) {
      bf16x16 Bf[8];
#pragma unroll
      for (int nt = 0; nt < 8; ++nt)
        Bf[nt] = load_frag_contig(W2s + (nt * 16 + nl) * 64 + kt * 32 + 16 * hl);
#pragma unroll
      for (int nt = 0; nt < 8; ++nt)
        acc[nt] = wmma_bf16(A2[kt], Bf[nt], acc[nt]);
    }

    // bias + relu; store C tile with columns in natural K order so that
    // layer-3 A fragments are 16 contiguous halves (2x ds_load_b128)
#pragma unroll
    for (int nt = 0; nt < 8; ++nt) {
      int N = nt * 16 + nl;
      int col = N;
      float bias = b2s[N];
#pragma unroll
      for (int r = 0; r < 8; ++r) {
        int M = r + 8 * hl;
        h2[M * 128 + col] = (__bf16)fmaxf(acc[nt][r] + bias, 0.f);
      }
    }

    // ---- layer 3: 16x128 @ 128x256 -> 64 WMMAs ----
    bf16x16 A3[4];
#pragma unroll
    for (int kt = 0; kt < 4; ++kt)
      A3[kt] = load_frag_contig(h2 + m * 128 + kt * 32 + 16 * hl);

#pragma unroll
    for (int nh = 0; nh < 2; ++nh) {
      f32x8 acc3[8];
#pragma unroll
      for (int nt = 0; nt < 8; ++nt) acc3[nt] = (f32x8)0.f;
#pragma unroll
      for (int kt = 0; kt < 4; ++kt) {
        bf16x16 Bf[8];
#pragma unroll
        for (int nt = 0; nt < 8; ++nt)
          Bf[nt] = load_frag_contig(
              W3s + (nh * 128 + nt * 16 + nl) * 128 + kt * 32 + 16 * hl);
#pragma unroll
        for (int nt = 0; nt < 8; ++nt)
          acc3[nt] = wmma_bf16(A3[kt], Bf[nt], acc3[nt]);
      }
      // max over the 16 points of this tile, then atomic max into vis
#pragma unroll
      for (int nt = 0; nt < 8; ++nt) {
        int N = nh * 128 + nt * 16 + nl;
        float mx = acc3[nt][0];
#pragma unroll
        for (int r = 1; r < 8; ++r) mx = fmaxf(mx, acc3[nt][r]);
        mx = fmaxf(mx, __shfl_xor(mx, 16, 32));
        mx += b3s[N];
        if (hl == 0) atomicMaxF(&vis[bt * 256 + N], mx);
      }
    }
  }
}

// ============================================================================
// Kernel 3: proprioception MLP 26->128->256 + LN  (256 rows)
// ============================================================================
__global__ __launch_bounds__(256) void prop_kernel(
    const float* prop_seq, const float* w1, const float* b1, const float* w2,
    const float* b2, const float* g, const float* be, float* prop_out) {
  __shared__ float in[26], hid[128], red[256];
  int row = blockIdx.x, tid = threadIdx.x;
  if (tid < 26) in[tid] = prop_seq[row * 26 + tid];
  __syncthreads();
  if (tid < 128) {
    float s = b1[tid];
    const float* w = w1 + tid * 26;
    for (int k = 0; k < 26; ++k) s += in[k] * w[k];
    hid[tid] = fmaxf(s, 0.f);
  }
  __syncthreads();
  float s = b2[tid];
  const float* w = w2 + tid * 128;
  for (int k = 0; k < 128; ++k) s += hid[k] * w[k];
  float mean = block_sum256(s, red) * (1.f / 256.f);
  float d = s - mean;
  float var = block_sum256(d * d, red) * (1.f / 256.f);
  prop_out[row * 256 + tid] = d * rsqrtf(var + EPS) * g[tid] + be[tid];
}

// ============================================================================
// Kernel 4: recurrent scan over T=4 (Sk=1 => softmax==1 => attn == out(v(kv)))
// ============================================================================
__global__ __launch_bounds__(256) void recurrent_kernel(
    const float* vis, const float* prop, const float* h_init,
    const float* ca_win, const float* ca_bin, const float* ca_wout,
    const float* ca_bout, const float* an_g, const float* an_b,
    const float* r_w1, const float* r_b1, const float* r_w2, const float* r_b2,
    const float* r_ln_g, const float* r_ln_b, const float* pos_emb,
    float* x_ws, float* h_final) {
  __shared__ float h[128], sv[256], sa[256], sf[256], red[256];
  int b = blockIdx.x, tid = threadIdx.x;
  if (tid < 128) h[tid] = h_init[tid];
  __syncthreads();
  for (int t = 0; t < 4; ++t) {
    const float* vrow = vis + (b * 4 + t) * 256;
    {  // value projection: rows 512..767 of ca_win
      const float* w = ca_win + (512 + tid) * 256;
      float s = ca_bin[512 + tid];
      for (int k = 0; k < 256; ++k) s += vrow[k] * w[k];
      sv[tid] = s;
    }
    __syncthreads();
    {  // out proj + prop residual
      const float* w = ca_wout + tid * 256;
      float s = ca_bout[tid];
      for (int k = 0; k < 256; ++k) s += sv[k] * w[k];
      sa[tid] = s + prop[(b * 4 + t) * 256 + tid];
    }
    __syncthreads();
    // LN -> fused; also emit transformer input (+pos_emb)
    float mean = block_sum256(sa[tid], red) * (1.f / 256.f);
    float d = sa[tid] - mean;
    float var = block_sum256(d * d, red) * (1.f / 256.f);
    float f = d * rsqrtf(var + EPS) * an_g[tid] + an_b[tid];
    sf[tid] = f;
    x_ws[(b * 4 + t) * 256 + tid] = f + pos_emb[t * 256 + tid];
    __syncthreads();
    // h update: 256->128 relu -> 128->128 -> LN(128)
    float hm = 0.f;
    if (tid < 128) {
      const float* w = r_w1 + tid * 256;
      float s = r_b1[tid];
      for (int k = 0; k < 256; ++k) s += sf[k] * w[k];
      hm = fmaxf(s, 0.f);
    }
    __syncthreads();
    if (tid < 128) sv[tid] = hm;
    __syncthreads();
    float hn = 0.f;
    if (tid < 128) {
      const float* w = r_w2 + tid * 128;
      float s = r_b2[tid];
      for (int k = 0; k < 128; ++k) s += sv[k] * w[k];
      hn = s;
    }
    float m2 = block_sum256(tid < 128 ? hn : 0.f, red) * (1.f / 128.f);
    float dd = (tid < 128) ? (hn - m2) : 0.f;
    float vv = block_sum256(dd * dd, red) * (1.f / 128.f);
    if (tid < 128) h[tid] = dd * rsqrtf(vv + EPS) * r_ln_g[tid] + r_ln_b[tid];
    __syncthreads();
  }
  if (tid < 128) h_final[b * 128 + tid] = h[tid];
}

// ============================================================================
// Kernel 5: one transformer layer on (B=64, T=4, E=256); launched 3x
// ============================================================================
__global__ __launch_bounds__(256) void xf_layer_kernel(
    float* x_ws, const float* ln1_g, const float* ln1_b, const float* sa_win,
    const float* sa_bin, const float* sa_wout, const float* sa_bout,
    const float* ln2_g, const float* ln2_b, const float* ff_w1,
    const float* ff_b1, const float* ff_w2, const float* ff_b2) {
  __shared__ float xs[1024], ys[1024], qkv[4 * 768], os[1024], ff[4 * 512],
      red[256];
  __shared__ float att[4][4][4];
  int b = blockIdx.x, tid = threadIdx.x;
  float* xrow = x_ws + b * 1024;
  for (int i = tid; i < 1024; i += 256) xs[i] = xrow[i];
  __syncthreads();
  for (int t = 0; t < 4; ++t) {  // LN1
    float v = xs[t * 256 + tid];
    float mean = block_sum256(v, red) * (1.f / 256.f);
    float d = v - mean;
    float var = block_sum256(d * d, red) * (1.f / 256.f);
    ys[t * 256 + tid] = d * rsqrtf(var + EPS) * ln1_g[tid] + ln1_b[tid];
  }
  __syncthreads();
  for (int i = tid; i < 4 * 768; i += 256) {  // QKV
    int t = i / 768, c = i % 768;
    const float* yr = ys + t * 256;
    const float* w = sa_win + c * 256;
    float s = sa_bin[c];
    for (int k = 0; k < 256; ++k) s += yr[k] * w[k];
    qkv[i] = s;
  }
  __syncthreads();
  if (tid < 16) {  // scores + softmax per (t, head)
    int t = tid >> 2, hh = tid & 3;
    float sc[4], mx = -1e30f;
    for (int k = 0; k < 4; ++k) {
      float s = 0.f;
      for (int d = 0; d < 64; ++d)
        s += qkv[t * 768 + hh * 64 + d] * qkv[k * 768 + 256 + hh * 64 + d];
      sc[k] = s * 0.125f;
      mx = fmaxf(mx, sc[k]);
    }
    float den = 0.f;
    for (int k = 0; k < 4; ++k) { sc[k] = __expf(sc[k] - mx); den += sc[k]; }
    for (int k = 0; k < 4; ++k) att[t][hh][k] = sc[k] / den;
  }
  __syncthreads();
  for (int i = tid; i < 1024; i += 256) {  // O = att @ V
    int t = i >> 8, c = i & 255, hh = c >> 6;
    float s = 0.f;
    for (int k = 0; k < 4; ++k) s += att[t][hh][k] * qkv[k * 768 + 512 + c];
    os[i] = s;
  }
  __syncthreads();
  for (int i = tid; i < 1024; i += 256) {  // out proj + residual
    int t = i >> 8, c = i & 255;
    const float* orow = os + t * 256;
    const float* w = sa_wout + c * 256;
    float s = sa_bout[c];
    for (int k = 0; k < 256; ++k) s += orow[k] * w[k];
    xs[i] += s;
  }
  __syncthreads();
  for (int t = 0; t < 4; ++t) {  // LN2
    float v = xs[t * 256 + tid];
    float mean = block_sum256(v, red) * (1.f / 256.f);
    float d = v - mean;
    float var = block_sum256(d * d, red) * (1.f / 256.f);
    ys[t * 256 + tid] = d * rsqrtf(var + EPS) * ln2_g[tid] + ln2_b[tid];
  }
  __syncthreads();
  for (int i = tid; i < 4 * 512; i += 256) {  // FF up
    int t = i >> 9, c = i & 511;
    const float* yr = ys + t * 256;
    const float* w = ff_w1 + c * 256;
    float s = ff_b1[c];
    for (int k = 0; k < 256; ++k) s += yr[k] * w[k];
    ff[i] = fmaxf(s, 0.f);
  }
  __syncthreads();
  for (int i = tid; i < 1024; i += 256) {  // FF down + residual
    int t = i >> 8, c = i & 255;
    const float* fr = ff + t * 512;
    const float* w = ff_w2 + c * 512;
    float s = ff_b2[c];
    for (int k = 0; k < 512; ++k) s += fr[k] * w[k];
    xs[i] += s;
  }
  __syncthreads();
  for (int i = tid; i < 1024; i += 256) xrow[i] = xs[i];
}

// ============================================================================
// Kernel 6: output heads (blocks 0..63: pred_z, 64..127: img)
// ============================================================================
__global__ __launch_bounds__(256) void heads_kernel(
    const float* x_ws, const float* h_final, const float* lh_w1,
    const float* lh_b1, const float* lh_w2, const float* lh_b2,
    const float* td_w1, const float* td_b1, const float* td_w2,
    const float* td_b2, float* out) {
  __shared__ float inb[512], hid[512];
  int tid = threadIdx.x;
  if (blockIdx.x < 64) {
    int b = blockIdx.x;
    inb[tid] = x_ws[(b * 4 + 3) * 256 + tid];
    __syncthreads();
    {
      const float* w = lh_w1 + tid * 256;
      float s = lh_b1[tid];
      for (int k = 0; k < 256; ++k) s += inb[k] * w[k];
      hid[tid] = fmaxf(s, 0.f);
    }
    __syncthreads();
    for (int i = tid; i < 320; i += 256) {
      const float* w = lh_w2 + i * 256;
      float s = lh_b2[i];
      for (int k = 0; k < 256; ++k) s += hid[k] * w[k];
      out[b * 320 + i] = s;
    }
  } else {
    int b = blockIdx.x - 64;
    if (tid < 128) inb[tid] = h_final[b * 128 + tid];
    __syncthreads();
    for (int i = tid; i < 512; i += 256) {
      const float* w = td_w1 + i * 128;
      float s = td_b1[i];
      for (int k = 0; k < 128; ++k) s += inb[k] * w[k];
      hid[i] = fmaxf(s, 0.f);
    }
    __syncthreads();
    for (int i = tid; i < 1440; i += 256) {
      const float* w = td_w2 + i * 512;
      float s = td_b2[i];
      for (int k = 0; k < 512; ++k) s += hid[k] * w[k];
      out[64 * 320 + b * 1440 + i] = s;
    }
  }
}

// ============================================================================
extern "C" void kernel_launch(void* const* d_in, const int* in_sizes, int n_in,
                              void* d_out, int out_size, void* d_ws,
                              size_t ws_size, hipStream_t stream) {
  (void)in_sizes; (void)n_in; (void)out_size; (void)ws_size;
#define FIN(i) ((const float*)d_in[(i)])
  char* ws = (char*)d_ws;
  __bf16* W2bf = (__bf16*)(ws + WS_W2BF);
  __bf16* W3bf = (__bf16*)(ws + WS_W3BF);
  float* W1f = (float*)(ws + WS_W1F);
  float* b1f = (float*)(ws + WS_B1F);
  float* b2f = (float*)(ws + WS_B2F);
  float* b3f = (float*)(ws + WS_B3F);
  float* vis = (float*)(ws + WS_VIS);
  float* prp = (float*)(ws + WS_PROPO);
  float* xws = (float*)(ws + WS_X);
  float* hfin = (float*)(ws + WS_HFIN);
  float* out = (float*)d_out;

  prep_kernel<<<256, 256, 0, stream>>>(
      FIN(IN_VW1), FIN(IN_VB1), FIN(IN_BN1_G), FIN(IN_BN1_B), FIN(IN_BN1_M),
      FIN(IN_BN1_V), FIN(IN_VW2), FIN(IN_VB2), FIN(IN_BN2_G), FIN(IN_BN2_B),
      FIN(IN_BN2_M), FIN(IN_BN2_V), FIN(IN_VW3), FIN(IN_VB3), FIN(IN_BN3_G),
      FIN(IN_BN3_B), FIN(IN_BN3_M), FIN(IN_BN3_V), W2bf, W3bf, W1f, b1f, b2f,
      b3f, vis);

  pointnet_kernel<<<dim3(256, 4), 256, 0, stream>>>(
      FIN(IN_PC), (const __bf16*)(ws + WS_W2BF), W1f, b1f, b2f, b3f, vis);

  prop_kernel<<<256, 256, 0, stream>>>(FIN(IN_PROP), FIN(IN_PM_W1),
                                       FIN(IN_PM_B1), FIN(IN_PM_W2),
                                       FIN(IN_PM_B2), FIN(IN_PM_LN_G),
                                       FIN(IN_PM_LN_B), prp);

  recurrent_kernel<<<64, 256, 0, stream>>>(
      vis, prp, FIN(IN_H_INIT), FIN(IN_CA_WIN), FIN(IN_CA_BIN),
      FIN(IN_CA_WOUT), FIN(IN_CA_BOUT), FIN(IN_AN_G), FIN(IN_AN_B),
      FIN(IN_R_W1), FIN(IN_R_B1), FIN(IN_R_W2), FIN(IN_R_B2), FIN(IN_R_LN_G),
      FIN(IN_R_LN_B), FIN(IN_POS_EMB), xws, hfin);

  for (int L = 0; L < 3; ++L) {
    int base = IN_L0 + L * 12;
    xf_layer_kernel<<<64, 256, 0, stream>>>(
        xws, FIN(base + 0), FIN(base + 1), FIN(base + 2), FIN(base + 3),
        FIN(base + 4), FIN(base + 5), FIN(base + 6), FIN(base + 7),
        FIN(base + 8), FIN(base + 9), FIN(base + 10), FIN(base + 11));
  }

  heads_kernel<<<128, 256, 0, stream>>>(
      xws, hfin, FIN(IN_LH_W1), FIN(IN_LH_B1), FIN(IN_LH_W2), FIN(IN_LH_B2),
      FIN(IN_TD_W1), FIN(IN_TD_B1), FIN(IN_TD_W2), FIN(IN_TD_B2), out);
#undef FIN
}
